// EncoderBlock_21053929685198
// MI455X (gfx1250) — matrix operand order repaired and checked by reading, not compiled, not run on hardware
//
#include <hip/hip_runtime.h>
#include <hip/hip_bf16.h>
#include <cstddef>
#include <cstdint>

// ---------------------------------------------------------------------------
// CDNA5 (gfx1250) encoder block: f16 WMMA GEMMs + flash attention (wave32).
// - All WMMA fragments are contiguous 16B LDS loads (ds_load_b128).
// - Fragments batch-preloaded so WMMAs issue back-to-back.
// - Row-major tiles staged with GLOBAL_LOAD_ASYNC_TO_LDS_B128 when the
//   toolchain exposes the builtin (ASYNCcnt path), manual pipeline otherwise.
// ---------------------------------------------------------------------------

typedef __attribute__((ext_vector_type(16))) _Float16 v16h;
typedef __attribute__((ext_vector_type(8)))  _Float16 v8h;
typedef __attribute__((ext_vector_type(8)))  float    v8f;
typedef int v4i __attribute__((vector_size(16)));

#define AS1 __attribute__((address_space(1)))
#define AS3 __attribute__((address_space(3)))

#if __has_builtin(__builtin_amdgcn_global_load_async_to_lds_b128) && \
    __has_builtin(__builtin_amdgcn_s_wait_asynccnt)
#define HAVE_ASYNC_LDS 1
#else
#define HAVE_ASYNC_LDS 0
#endif

constexpr int Bc   = 2;
constexpr int Tc   = 2048;
constexpr int Dc   = 1024;
constexpr int Hc   = 16;
constexpr int KVHc = 4;
constexpr int HDc  = 64;
constexpr int DFFc = 4096;
constexpr int KWc  = 15;
constexpr int NTc  = Bc * Tc;          // 4096 token rows
constexpr float EPSc = 1e-6f;

static __device__ __forceinline__ v16h make16(v8h lo, v8h hi) {
  return __builtin_shufflevector(lo, hi, 0, 1, 2, 3, 4, 5, 6, 7,
                                         8, 9, 10, 11, 12, 13, 14, 15);
}

#if HAVE_ASYNC_LDS
static __device__ __forceinline__ void async_b128(const _Float16* g, _Float16* lds) {
  __builtin_amdgcn_global_load_async_to_lds_b128(
      (AS1 v4i*)const_cast<_Float16*>(g), (AS3 v4i*)lds, 0, 0);
}
#endif

// ---------------------------------------------------------------------------
// Elementwise helpers
// ---------------------------------------------------------------------------
__global__ void k_cvt_f32_f16(const float* __restrict__ X, _Float16* __restrict__ Y, int n) {
  int i = blockIdx.x * 256 + threadIdx.x;
  if (i < n) Y[i] = (_Float16)X[i];
}

__global__ void k_relu_mul_f16(const float* __restrict__ A, const float* __restrict__ Bv,
                               _Float16* __restrict__ Y, int n) {
  int i = blockIdx.x * 256 + threadIdx.x;
  if (i < n) Y[i] = (_Float16)(fmaxf(A[i], 0.0f) * fmaxf(Bv[i], 0.0f));
}

__global__ void k_glu(const float* __restrict__ U, float* __restrict__ G, int n) {
  int i = blockIdx.x * 256 + threadIdx.x;
  if (i >= n) return;
  int r = i / Dc, c = i % Dc;
  float u1 = U[(size_t)r * (2 * Dc) + c];
  float u2 = U[(size_t)r * (2 * Dc) + Dc + c];
  G[i] = u1 * (1.0f / (1.0f + __expf(-u2)));
}

__global__ void k_dwconv(const float* __restrict__ G, const float* __restrict__ Wd,
                         float* __restrict__ C, int n) {
  int i = blockIdx.x * 256 + threadIdx.x;
  if (i >= n) return;
  int d = i % Dc;
  int t = (i / Dc) % Tc;
  int b = i / (Dc * Tc);
  float acc = 0.0f;
  #pragma unroll
  for (int k = 0; k < KWc; ++k) {
    int tt = t + k - KWc / 2;
    if (tt >= 0 && tt < Tc) acc += G[((size_t)b * Tc + tt) * Dc + d] * Wd[k * Dc + d];
  }
  C[i] = acc;
}

// ---------------------------------------------------------------------------
// Row RMS-norm (zcrms): fp32 in, f16 out.  One block per row of length Dl.
// ---------------------------------------------------------------------------
__global__ __launch_bounds__(256) void k_rmsnorm_f16(const float* __restrict__ X,
                                                     const float* __restrict__ scale,
                                                     _Float16* __restrict__ Y, int Dl) {
  const int row = blockIdx.x;
  const float* xr = X + (size_t)row * Dl;
  float ss = 0.0f;
  for (int d = threadIdx.x; d < Dl; d += 256) { float v = xr[d]; ss += v * v; }
  for (int m = 1; m < 32; m <<= 1) ss += __shfl_xor(ss, m, 32);
  __shared__ float wsum[8];
  if ((threadIdx.x & 31) == 0) wsum[threadIdx.x >> 5] = ss;
  __syncthreads();
  float tot = 0.0f;
  #pragma unroll
  for (int i = 0; i < 8; ++i) tot += wsum[i];
  const float rms = sqrtf(tot / (float)Dl + EPSc);
  _Float16* yr = Y + (size_t)row * Dl;
  for (int d = threadIdx.x; d < Dl; d += 256)
    yr[d] = (_Float16)((1.0f + scale[d]) * xr[d] / rms);
}

// ---------------------------------------------------------------------------
// Per-head RMS-norm + RoPE + transpose [B,T,heads,HD] -> f16 [B,heads,T,HD].
// One wave32 per row of HD=64 (2 elems/lane: d and d+32).
// ---------------------------------------------------------------------------
__global__ __launch_bounds__(256) void k_head_prep(const float* __restrict__ X,
                                                   _Float16* __restrict__ Y,
                                                   const float* __restrict__ scale,
                                                   const float* __restrict__ cosT,
                                                   const float* __restrict__ sinT,
                                                   int heads, int rope, int rows) {
  const int lane = threadIdx.x & 31;
  const int wv   = threadIdx.x >> 5;
  const int row  = blockIdx.x * 8 + wv;
  if (row >= rows) return;
  const int b = row / (Tc * heads);
  const int t = (row / heads) % Tc;
  const int h = row % heads;
  const float* xr = X + (size_t)row * HDc;
  float x1 = xr[lane], x2 = xr[lane + 32];
  float n1 = x1, n2 = x2;
  if (scale) {
    float ss = x1 * x1 + x2 * x2;
    for (int m = 1; m < 32; m <<= 1) ss += __shfl_xor(ss, m, 32);
    const float rms = sqrtf(ss * (1.0f / (float)HDc) + EPSc);
    n1 = (1.0f + scale[lane])      * x1 / rms;
    n2 = (1.0f + scale[lane + 32]) * x2 / rms;
  }
  float o1 = n1, o2 = n2;
  if (rope) {
    const float c = cosT[t * (HDc / 2) + lane];
    const float s = sinT[t * (HDc / 2) + lane];
    o1 = n1 * c - n2 * s;
    o2 = n2 * c + n1 * s;
  }
  _Float16* yr = Y + ((size_t)(b * heads + h) * Tc + t) * HDc;
  yr[lane]      = (_Float16)o1;
  yr[lane + 32] = (_Float16)o2;
}

// ---------------------------------------------------------------------------
// WMMA GEMM: C[M,N] (f32) = A[M,K] (f16) x B[K,N] (f16) (+ optional residual).
// Block 256 thr = 8 waves; tile 128x64, BK=64 (2 WMMA k-steps per stage).
// A staged row-major (async DMA when available); B staged TRANSPOSED so every
// lane's B-fragment is two contiguous 16B LDS loads.  M%128, N%64, K%64 == 0.
// ---------------------------------------------------------------------------
__global__ __launch_bounds__(256) void k_wmma_gemm(const _Float16* __restrict__ A,
                                                   const _Float16* __restrict__ Bw,
                                                   float* __restrict__ C,
                                                   const float* __restrict__ Res,
                                                   int M, int N, int K) {
  constexpr int BM = 128, BN = 64, BK = 64;
  __shared__ _Float16 As [BM][BK + 8];   // row stride 72 halves = 144B (16B mult)
  __shared__ _Float16 BsT[BN][BK + 8];   // transposed: [n][k]
  const int tid  = threadIdx.x;
  const int lane = tid & 31;
  const int wv   = tid >> 5;      // wave id 0..7
  const int hv   = lane >> 4;     // half-wave: 0 or 1
  const int l15  = lane & 15;
  const int tm   = blockIdx.y * BM;
  const int tn   = blockIdx.x * BN;

  v8f acc[4];
  #pragma unroll
  for (int nc = 0; nc < 4; ++nc)
    #pragma unroll
    for (int j = 0; j < 8; ++j) acc[nc][j] = 0.0f;

  // staging coordinates
  const int ar  = tid >> 1;            // A row 0..127
  const int ac0 = (tid & 1) * 32;      // A col block: 32 halves
  const int br  = tid >> 2;            // B k-row 0..63
  const int bn0 = (tid & 3) * 16;      // B n block: 16 halves

  for (int k0 = 0; k0 < K; k0 += BK) {
    __syncthreads();
    // ---- stage A tile (row-major) ----
#if HAVE_ASYNC_LDS
    {
      #pragma unroll
      for (int j = 0; j < 4; ++j)
        async_b128(&A[(size_t)(tm + ar) * K + k0 + ac0 + 8 * j], &As[ar][ac0 + 8 * j]);
    }
#else
    {
      const v8h* src = (const v8h*)&A[(size_t)(tm + ar) * K + k0 + ac0];
      v8h a0 = src[0], a1 = src[1], a2 = src[2], a3 = src[3];
      *(v8h*)&As[ar][ac0]      = a0;
      *(v8h*)&As[ar][ac0 + 8]  = a1;
      *(v8h*)&As[ar][ac0 + 16] = a2;
      *(v8h*)&As[ar][ac0 + 24] = a3;
    }
#endif
    // ---- stage B tile transposed ----
    {
      const v8h* src = (const v8h*)&Bw[(size_t)(k0 + br) * N + tn + bn0];
      v8h b0 = src[0], b1 = src[1];
      #pragma unroll
      for (int j = 0; j < 8; ++j) BsT[bn0 + j][br]     = b0[j];
      #pragma unroll
      for (int j = 0; j < 8; ++j) BsT[bn0 + 8 + j][br] = b1[j];
    }
    if (k0 + BK < K) {   // global_prefetch_b8 of next tiles
      __builtin_prefetch(&A[(size_t)(tm + ar) * K + (k0 + BK) + ac0], 0, 1);
      __builtin_prefetch(&Bw[(size_t)(k0 + BK + br) * N + tn + bn0], 0, 1);
    }
#if HAVE_ASYNC_LDS
    __builtin_amdgcn_s_wait_asynccnt(0);
#endif
    __syncthreads();

    const int arow = wv * 16 + l15;
    const int kb   = hv * 8;
    #pragma unroll
    for (int kk = 0; kk < 2; ++kk) {
      // batch-preload A fragment + all 4 B fragments, then burst 4 WMMAs
      v8h alo = *(const v8h*)&As[arow][kk * 32 + kb];
      v8h ahi = *(const v8h*)&As[arow][kk * 32 + 16 + kb];
      v16h af = make16(alo, ahi);
      v16h bfr[4];
      #pragma unroll
      for (int nc = 0; nc < 4; ++nc) {
        const int col = nc * 16 + l15;
        v8h blo = *(const v8h*)&BsT[col][kk * 32 + hv * 8];
        v8h bhi = *(const v8h*)&BsT[col][kk * 32 + 16 + hv * 8];
        bfr[nc] = make16(blo, bhi);
      }
      #pragma unroll
      for (int nc = 0; nc < 4; ++nc)
        acc[nc] = __builtin_amdgcn_wmma_f32_16x16x32_f16(
            false, af, false, bfr[nc], (short)0, acc[nc], false, false);
    }
  }

  // C layout: VGPR v -> row M = v + 8*hv, lane col = l15
  #pragma unroll
  for (int nc = 0; nc < 4; ++nc) {
    const int col = tn + nc * 16 + l15;
    #pragma unroll
    for (int v = 0; v < 8; ++v) {
      const int row = tm + wv * 16 + hv * 8 + v;
      float val = acc[nc][v];
      if (Res) val += Res[(size_t)row * N + col];
      C[(size_t)row * N + col] = val;
    }
  }
}

// ---------------------------------------------------------------------------
// Flash attention (non-causal, mask==all-true): one wave per 16 query rows.
// Block = 8 waves (128 q rows); loop over key blocks of 32.  K staged
// row-major (async DMA when available), V staged TRANSPOSED.  QK^T and PV
// via WMMA f16 (HD=64 -> 2 chained K32 steps).  Output f32 [B, T, H*HD].
// ---------------------------------------------------------------------------
__global__ __launch_bounds__(256) void k_attention(const _Float16* __restrict__ Q,
                                                   const _Float16* __restrict__ Km,
                                                   const _Float16* __restrict__ Vm,
                                                   float* __restrict__ O) {
  __shared__ _Float16 Ks [32][HDc + 8];   // [key][d] row stride 72h = 144B
  __shared__ _Float16 VsT[HDc][32 + 8];   // [d][key] row stride 40h = 80B
  __shared__ _Float16 Ps[8][16][40];      // per-wave P staging (16 rows x 32 keys)

  const int tid  = threadIdx.x;
  const int lane = tid & 31;
  const int wv   = tid >> 5;
  const int hv   = lane >> 4;
  const int l15  = lane & 15;

  const int bh = blockIdx.y;
  const int bb = bh / Hc;
  const int hh = bh % Hc;
  const int kv = hh / (Hc / KVHc);

  const _Float16* Qp = Q  + (size_t)(bb * Hc   + hh) * Tc * HDc;
  const _Float16* Kp = Km + (size_t)(bb * KVHc + kv) * Tc * HDc;
  const _Float16* Vp = Vm + (size_t)(bb * KVHc + kv) * Tc * HDc;

  const int qbase = blockIdx.x * 128 + wv * 16;

  // Q fragments (A layout) for contraction chunks K=[0,32) and [32,64)
  v16h qf[2];
  #pragma unroll
  for (int kk = 0; kk < 2; ++kk) {
    v8h qlo = *(const v8h*)&Qp[(size_t)(qbase + l15) * HDc + kk * 32 + hv * 8];
    v8h qhi = *(const v8h*)&Qp[(size_t)(qbase + l15) * HDc + kk * 32 + 16 + hv * 8];
    qf[kk] = make16(qlo, qhi);
  }

  float mrow[8], lsum[8], sf[8];
  v8f o[4];
  #pragma unroll
  for (int v = 0; v < 8; ++v) { mrow[v] = -3.0e38f; lsum[v] = 0.0f; }
  #pragma unroll
  for (int nc = 0; nc < 4; ++nc)
    #pragma unroll
    for (int v = 0; v < 8; ++v) o[nc][v] = 0.0f;

  // staging coords: 256 threads cover 32 rows x 8 v8h-chunks exactly
  const int sr  = tid >> 3;
  const int sc0 = (tid & 7) * 8;

  for (int kb0 = 0; kb0 < Tc; kb0 += 32) {
    __syncthreads();
    {
#if HAVE_ASYNC_LDS
      async_b128(&Kp[(size_t)(kb0 + sr) * HDc + sc0], &Ks[sr][sc0]);
      v8h vv = *(const v8h*)&Vp[(size_t)(kb0 + sr) * HDc + sc0];
      #pragma unroll
      for (int j = 0; j < 8; ++j) VsT[sc0 + j][sr] = vv[j];
      __builtin_amdgcn_s_wait_asynccnt(0);
#else
      v8h kk8 = *(const v8h*)&Kp[(size_t)(kb0 + sr) * HDc + sc0];
      v8h vv  = *(const v8h*)&Vp[(size_t)(kb0 + sr) * HDc + sc0];
      *(v8h*)&Ks[sr][sc0] = kk8;
      #pragma unroll
      for (int j = 0; j < 8; ++j) VsT[sc0 + j][sr] = vv[j];
#endif
    }
    __syncthreads();

    // S = Q K^T for 32 keys: preload all 4 K^T fragments, then 4 WMMAs
    v16h kfr[2][2];
    #pragma unroll
    for (int j = 0; j < 2; ++j)
      #pragma unroll
      for (int kk = 0; kk < 2; ++kk) {
        v8h blo = *(const v8h*)&Ks[j * 16 + l15][kk * 32 + hv * 8];
        v8h bhi = *(const v8h*)&Ks[j * 16 + l15][kk * 32 + 16 + hv * 8];
        kfr[j][kk] = make16(blo, bhi);
      }
    v8f s[2];
    #pragma unroll
    for (int j = 0; j < 2; ++j) {
      #pragma unroll
      for (int v = 0; v < 8; ++v) s[j][v] = 0.0f;
      #pragma unroll
      for (int kk = 0; kk < 2; ++kk)
        s[j] = __builtin_amdgcn_wmma_f32_16x16x32_f16(
            false, qf[kk], false, kfr[j][kk], (short)0, s[j], false, false);
    }

    // Online softmax per row (row M spread across the 16-lane half-group)
    #pragma unroll
    for (int v = 0; v < 8; ++v) {
      float a0 = s[0][v] * 0.125f;   // 1/sqrt(HD)
      float a1 = s[1][v] * 0.125f;
      float r = fmaxf(a0, a1);
      for (int m = 1; m < 16; m <<= 1) r = fmaxf(r, __shfl_xor(r, m, 32));
      const float mn = fmaxf(mrow[v], r);
      sf[v] = __expf(mrow[v] - mn);
      mrow[v] = mn;
      const float p0 = __expf(a0 - mn);
      const float p1 = __expf(a1 - mn);
      float rs = p0 + p1;
      for (int m = 1; m < 16; m <<= 1) rs += __shfl_xor(rs, m, 32);
      lsum[v] = lsum[v] * sf[v] + rs;
      Ps[wv][hv * 8 + v][l15]      = (_Float16)p0;
      Ps[wv][hv * 8 + v][16 + l15] = (_Float16)p1;
    }
    __syncthreads();

    // rescale running O
    #pragma unroll
    for (int nc = 0; nc < 4; ++nc)
      #pragma unroll
      for (int v = 0; v < 8; ++v) o[nc][v] *= sf[v];

    // P fragment (A layout) + preload 4 V fragments, then 4 WMMAs
    v8h plo = *(const v8h*)&Ps[wv][l15][hv * 8];
    v8h phi = *(const v8h*)&Ps[wv][l15][16 + hv * 8];
    v16h pf = make16(plo, phi);
    v16h vfr[4];
    #pragma unroll
    for (int nc = 0; nc < 4; ++nc) {
      v8h blo = *(const v8h*)&VsT[nc * 16 + l15][hv * 8];
      v8h bhi = *(const v8h*)&VsT[nc * 16 + l15][16 + hv * 8];
      vfr[nc] = make16(blo, bhi);
    }
    #pragma unroll
    for (int nc = 0; nc < 4; ++nc)
      o[nc] = __builtin_amdgcn_wmma_f32_16x16x32_f16(
          false, pf, false, vfr[nc], (short)0, o[nc], false, false);
  }

  #pragma unroll
  for (int nc = 0; nc < 4; ++nc) {
    #pragma unroll
    for (int v = 0; v < 8; ++v) {
      const int trow = qbase + hv * 8 + v;
      O[((size_t)(bb * Tc + trow) * Hc + hh) * HDc + nc * 16 + l15] = o[nc][v] / lsum[v];
    }
  }
}

// ---------------------------------------------------------------------------
// Host orchestration
// ---------------------------------------------------------------------------
extern "C" void kernel_launch(void* const* d_in, const int* in_sizes, int n_in,
                              void* d_out, int out_size, void* d_ws, size_t ws_size,
                              hipStream_t stream) {
  (void)in_sizes; (void)n_in; (void)out_size; (void)ws_size;

  const float* x        = (const float*)d_in[0];
  const float* cosT     = (const float*)d_in[2];
  const float* sinT     = (const float*)d_in[3];
  const float* attn_ns  = (const float*)d_in[4];
  const float* Wq       = (const float*)d_in[5];
  const float* Wk       = (const float*)d_in[6];
  const float* Wv       = (const float*)d_in[7];
  const float* q_ns     = (const float*)d_in[8];
  const float* k_ns     = (const float*)d_in[9];
  const float* Wo       = (const float*)d_in[10];
  const float* conv_ns  = (const float*)d_in[11];
  const float* Wup      = (const float*)d_in[12];
  const float* Wdw      = (const float*)d_in[13];
  const float* inner_ns = (const float*)d_in[14];
  const float* Wpd      = (const float*)d_in[15];
  const float* ffn_ns   = (const float*)d_in[16];
  const float* Wg       = (const float*)d_in[17];
  const float* Wu       = (const float*)d_in[18];
  const float* Wd       = (const float*)d_in[19];
  float* out            = (float*)d_out;

  char* ws = (char*)d_ws;
  size_t off = 0;
  auto allocH = [&](size_t n) { _Float16* p = (_Float16*)(ws + off);
                                off += ((n * 2 + 255) / 256) * 256; return p; };
  auto allocF = [&](size_t n) { float* p = (float*)(ws + off);
                                off += ((n * 4 + 255) / 256) * 256; return p; };

  // f16 weights
  _Float16* Wq16  = allocH((size_t)Dc * Dc);
  _Float16* Wk16  = allocH((size_t)Dc * KVHc * HDc);
  _Float16* Wv16  = allocH((size_t)Dc * KVHc * HDc);
  _Float16* Wo16  = allocH((size_t)Dc * Dc);
  _Float16* Wup16 = allocH((size_t)Dc * 2 * Dc);
  _Float16* Wpd16 = allocH((size_t)Dc * Dc);
  _Float16* Wg16  = allocH((size_t)Dc * DFFc);
  _Float16* Wu16  = allocH((size_t)Dc * DFFc);
  _Float16* Wd16  = allocH((size_t)DFFc * Dc);
  // activations
  _Float16* h16    = allocH((size_t)NTc * Dc);        // normalized acts (reused)
  float*    qf     = allocF((size_t)NTc * Dc);
  float*    kf     = allocF((size_t)NTc * KVHc * HDc);
  float*    vf     = allocF((size_t)NTc * KVHc * HDc);
  _Float16* q16    = allocH((size_t)NTc * Dc);        // [B,H,T,HD]
  _Float16* k16    = allocH((size_t)Bc * KVHc * Tc * HDc);
  _Float16* v16    = allocH((size_t)Bc * KVHc * Tc * HDc);
  float*    attnf  = allocF((size_t)NTc * Dc);
  _Float16* attn16 = allocH((size_t)NTc * Dc);
  float*    x1     = allocF((size_t)NTc * Dc);
  float*    u      = allocF((size_t)NTc * 2 * Dc);
  float*    gbuf   = allocF((size_t)NTc * Dc);
  float*    cbuf   = allocF((size_t)NTc * Dc);
  float*    x2     = allocF((size_t)NTc * Dc);
  float*    gatef  = allocF((size_t)NTc * DFFc);
  float*    upf    = allocF((size_t)NTc * DFFc);
  _Float16* ff16   = allocH((size_t)NTc * DFFc);

  auto cvt = [&](const float* src, _Float16* dst, int n) {
    k_cvt_f32_f16<<<(n + 255) / 256, 256, 0, stream>>>(src, dst, n);
  };
  auto gemm = [&](const _Float16* A, const _Float16* Bm, float* C, const float* R,
                  int M, int N, int K) {
    dim3 grid(N / 64, M / 128);
    k_wmma_gemm<<<grid, 256, 0, stream>>>(A, Bm, C, R, M, N, K);
  };

  // weight conversion
  cvt(Wq,  Wq16,  Dc * Dc);
  cvt(Wk,  Wk16,  Dc * KVHc * HDc);
  cvt(Wv,  Wv16,  Dc * KVHc * HDc);
  cvt(Wo,  Wo16,  Dc * Dc);
  cvt(Wup, Wup16, Dc * 2 * Dc);
  cvt(Wpd, Wpd16, Dc * Dc);
  cvt(Wg,  Wg16,  Dc * DFFc);
  cvt(Wu,  Wu16,  Dc * DFFc);
  cvt(Wd,  Wd16,  DFFc * Dc);

  // ---- attention block ----
  k_rmsnorm_f16<<<NTc, 256, 0, stream>>>(x, attn_ns, h16, Dc);
  gemm(h16, Wq16, qf, nullptr, NTc, Dc, Dc);
  gemm(h16, Wk16, kf, nullptr, NTc, KVHc * HDc, Dc);
  gemm(h16, Wv16, vf, nullptr, NTc, KVHc * HDc, Dc);

  k_head_prep<<<(NTc * Hc) / 8, 256, 0, stream>>>(qf, q16, q_ns, cosT, sinT, Hc, 1, NTc * Hc);
  k_head_prep<<<(NTc * KVHc) / 8, 256, 0, stream>>>(kf, k16, k_ns, cosT, sinT, KVHc, 1, NTc * KVHc);
  k_head_prep<<<(NTc * KVHc) / 8, 256, 0, stream>>>(vf, v16, nullptr, nullptr, nullptr, KVHc, 0, NTc * KVHc);

  {
    dim3 grid(Tc / 128, Bc * Hc);
    k_attention<<<grid, 256, 0, stream>>>(q16, k16, v16, attnf);
  }
  cvt(attnf, attn16, NTc * Dc);
  gemm(attn16, Wo16, x1, x, NTc, Dc, Dc);            // x1 = x + attn @ Wo

  // ---- conv block ----
  k_rmsnorm_f16<<<NTc, 256, 0, stream>>>(x1, conv_ns, h16, Dc);
  gemm(h16, Wup16, u, nullptr, NTc, 2 * Dc, Dc);
  k_glu<<<(NTc * Dc + 255) / 256, 256, 0, stream>>>(u, gbuf, NTc * Dc);
  k_dwconv<<<(NTc * Dc + 255) / 256, 256, 0, stream>>>(gbuf, Wdw, cbuf, NTc * Dc);
  k_rmsnorm_f16<<<NTc, 256, 0, stream>>>(cbuf, inner_ns, h16, Dc);
  gemm(h16, Wpd16, x2, x1, NTc, Dc, Dc);             // x2 = x1 + c @ Wpd

  // ---- FFN block ----
  k_rmsnorm_f16<<<NTc, 256, 0, stream>>>(x2, ffn_ns, h16, Dc);
  gemm(h16, Wg16, gatef, nullptr, NTc, DFFc, Dc);
  gemm(h16, Wu16, upf,   nullptr, NTc, DFFc, Dc);
  k_relu_mul_f16<<<(NTc * DFFc + 255) / 256, 256, 0, stream>>>(gatef, upf, ff16, NTc * DFFc);
  gemm(ff16, Wd16, out, x2, NTc, Dc, DFFc);          // out = x2 + ff @ Wd
}
